// MultiModalAttention_627065225467
// MI455X (gfx1250) — matrix operand order repaired and checked by reading, not compiled, not run on hardware
//
#include <hip/hip_runtime.h>
#include <hip/hip_bf16.h>
#include <stdint.h>

// ---------------------------------------------------------------------------
// Types for CDNA5 WMMA (wave32): v_wmma_f32_16x16x32_bf16
// ---------------------------------------------------------------------------
typedef __attribute__((ext_vector_type(8)))  float        v8f;
typedef __attribute__((ext_vector_type(16))) __bf16       v16bf;
typedef __attribute__((ext_vector_type(4)))  unsigned int u32x4;
typedef __attribute__((ext_vector_type(4)))  int          i32x4;

union ABTile { u32x4 u4[2]; v16bf v; };
static_assert(sizeof(v16bf) == 32, "v16bf must be 8 VGPRs");

// ---------------------------------------------------------------------------
// CDNA5 async global->LDS copy (ASYNCcnt path), with portable fallback.
// Builtin signature (from hipcc diagnostic): first arg is int4 AS(1)*,
// second is int4 AS(3)*, then two int immediates (offset, cpol).
// ---------------------------------------------------------------------------
#if defined(__has_builtin)
#if __has_builtin(__builtin_amdgcn_global_load_async_to_lds_b128)
#define HAVE_ASYNC_CP 1
#endif
#endif
#ifndef HAVE_ASYNC_CP
#define HAVE_ASYNC_CP 0
#endif

#if HAVE_ASYNC_CP
typedef __attribute__((address_space(1))) i32x4 GI4;  // global int4
typedef __attribute__((address_space(3))) i32x4 LI4;  // LDS int4
#if __has_builtin(__builtin_amdgcn_s_wait_asynccnt)
#define ASYNC_WAIT(n) __builtin_amdgcn_s_wait_asynccnt(n)
#else
#define ASYNC_WAIT(n) asm volatile("s_wait_asynccnt %0" ::"i"(n) : "memory")
#endif
#else
#define ASYNC_WAIT(n)
#endif

__device__ __forceinline__ void cp_g2l_16B(const unsigned short* g,
                                           unsigned short* l) {
#if HAVE_ASYNC_CP
  __builtin_amdgcn_global_load_async_to_lds_b128((GI4*)g, (LI4*)l, 0, 0);
#else
  *(u32x4*)l = *(const u32x4*)g;
#endif
}

__device__ __forceinline__ unsigned short f2bf(float f) {
  unsigned int u = __float_as_uint(f);
  u += 0x7fffu + ((u >> 16) & 1u);  // round-to-nearest-even
  return (unsigned short)(u >> 16);
}

// A operand 16x32 bf16: lane L -> row (L%16); K = {0..7,16..23} for lanes 0-15,
// {8..15,24..31} for lanes 16-31  (ISA 7.12.2, 16-bit A layout).
__device__ __forceinline__ v16bf load_a16x32(const unsigned short* base, int ld,
                                             int row, int k0, int hi) {
  const unsigned short* p = base + (size_t)row * ld + k0 + hi * 8;
  ABTile t;
  t.u4[0] = *(const u32x4*)(p);       // K = k0+off .. +7
  t.u4[1] = *(const u32x4*)(p + 16);  // K = k0+off+16 .. +23
  return t.v;
}

// B operand 32x16 bf16 built from rows of a row-major [N,K] matrix (columns of
// B == W rows): lane L -> column (L%16); K = {0..7,16..23} for lanes 0-15,
// {8..15,24..31} for lanes 16-31 (same order as A).
__device__ __forceinline__ v16bf load_b32x16(const unsigned short* base, int ld,
                                             int row, int k0, int hi) {
  const unsigned short* p = base + (size_t)row * ld + k0 + hi * 8;
  ABTile t;
  t.u4[0] = *(const u32x4*)(p);
  t.u4[1] = *(const u32x4*)(p + 16);
  return t.v;
}

#define WMMA_BF16(a, b, c) \
  __builtin_amdgcn_wmma_f32_16x16x32_bf16(false, (a), false, (b), (short)0, (c), false, false)

// ---------------------------------------------------------------------------
// f32 -> bf16 cast (scale folding used to fold SCALE into Wq)
// ---------------------------------------------------------------------------
__global__ void cvt_f32_bf16(const float* __restrict__ src,
                             unsigned short* __restrict__ dst,
                             size_t n, float scale) {
  size_t i = (size_t)blockIdx.x * blockDim.x + threadIdx.x;
  size_t stride = (size_t)gridDim.x * blockDim.x;
  for (; i < n; i += stride) dst[i] = f2bf(src[i] * scale);
}

// ---------------------------------------------------------------------------
// bf16 GEMM: C[M,N] = A[M,K] @ W[N,K]^T
// Block = 256 threads = 8 waves (4x2). Block tile 128x64, k-step 32.
// A/B tiles double-buffered in LDS via async global->LDS copies (3 async
// instructions per wave per stage; in-order completion => s_wait_asynccnt 3
// overlaps the next stage's copy with this stage's WMMAs).
// LDS row pitch = 40 elems (80B): 16B-aligned b128 reads, staggered banks.
// ---------------------------------------------------------------------------
template <bool F32OUT>
__global__ __launch_bounds__(256) void gemm_bf16_wmma(
    const unsigned short* __restrict__ A, const unsigned short* __restrict__ W,
    void* __restrict__ Cout, const float* __restrict__ bias,
    int K, int lda, int ldb, int ldc) {
  constexpr int LDT = 40;
  __shared__ __align__(16) unsigned short As[2][128 * LDT];  // 2 x 10KB
  __shared__ __align__(16) unsigned short Bs[2][64 * LDT];   // 2 x 5KB

  const int t = threadIdx.x;
  const int lane = t & 31, wave = t >> 5;
  const int l16 = lane & 15, hi = lane >> 4;
  const int bm = blockIdx.x * 128, bn = blockIdx.y * 64;
  const int wm = (wave >> 1) * 32, wn = (wave & 1) * 32;

  // copy assignment: 16B chunks; A has 512 chunks (2/thread), B has 256.
  const int arow = t >> 2;           // 0..63
  const int koff = (t & 3) * 8;      // element offset within 32-wide k-slab
  const int brow = t >> 2;           // 0..63

  auto stage = [&](int buf, int k0) {
    cp_g2l_16B(A + (size_t)(bm + arow) * lda + k0 + koff,
               &As[buf][arow * LDT + koff]);
    cp_g2l_16B(A + (size_t)(bm + arow + 64) * lda + k0 + koff,
               &As[buf][(arow + 64) * LDT + koff]);
    cp_g2l_16B(W + (size_t)(bn + brow) * ldb + k0 + koff,
               &Bs[buf][brow * LDT + koff]);
  };

  v8f acc[2][2] = {};
  stage(0, 0);
  for (int k0 = 0; k0 < K; k0 += 32) {
    const int cur = (k0 >> 5) & 1;
    if (k0 + 32 < K) {
      stage(cur ^ 1, k0 + 32);  // prefetch next slab into other buffer
      ASYNC_WAIT(3);            // wait for *this* slab; next stays in flight
    } else {
      ASYNC_WAIT(0);
    }
    __syncthreads();
    v16bf a0 = load_a16x32(&As[cur][0], LDT, wm + l16,      0, hi);
    v16bf a1 = load_a16x32(&As[cur][0], LDT, wm + 16 + l16, 0, hi);
    v16bf b0 = load_b32x16(&Bs[cur][0], LDT, wn + l16,      0, hi);
    v16bf b1 = load_b32x16(&Bs[cur][0], LDT, wn + 16 + l16, 0, hi);
    acc[0][0] = WMMA_BF16(a0, b0, acc[0][0]);
    acc[0][1] = WMMA_BF16(a0, b1, acc[0][1]);
    acc[1][0] = WMMA_BF16(a1, b0, acc[1][0]);
    acc[1][1] = WMMA_BF16(a1, b1, acc[1][1]);
    __syncthreads();  // all waves done reading before buffer is re-staged
  }

  // C/D layout: VGPR r -> row r + 8*hi ; lane%16 -> column
#pragma unroll
  for (int im = 0; im < 2; ++im)
#pragma unroll
    for (int in = 0; in < 2; ++in)
#pragma unroll
      for (int r = 0; r < 8; ++r) {
        int row = bm + wm + im * 16 + r + hi * 8;
        int col = bn + wn + in * 16 + l16;
        float val = acc[im][in][r];
        if (F32OUT)
          ((float*)Cout)[(size_t)row * ldc + col] = val + bias[col];
        else
          ((unsigned short*)Cout)[(size_t)row * ldc + col] = f2bf(val);
      }
}

// ---------------------------------------------------------------------------
// V transpose per head: vt[b,h,d,j] = kv[b,j, 1024 + h*64 + d]
// ---------------------------------------------------------------------------
__global__ void vtrans_kernel(const unsigned short* __restrict__ kvb,
                              unsigned short* __restrict__ vtb) {
  size_t i = (size_t)blockIdx.x * blockDim.x + threadIdx.x;  // < 4*16*64*1024
  int j = (int)(i & 1023);
  int d = (int)((i >> 10) & 63);
  int h = (int)((i >> 16) & 15);
  int b = (int)(i >> 20);
  vtb[i] = kvb[((size_t)(b * 1024 + j)) * 2048 + 1024 + h * 64 + d];
}

// ---------------------------------------------------------------------------
// Flash attention, one wave (32 lanes) per (b, h, 32-row Q tile).
// Q pre-scaled (SCALE folded into Wq). Online softmax in f32.
// ---------------------------------------------------------------------------
__global__ __launch_bounds__(32) void attn_fwd_wmma(
    const unsigned short* __restrict__ qb,    // [B*2048,1024] bf16
    const unsigned short* __restrict__ kvb,   // [B*1024,2048] bf16 (K | V)
    const unsigned short* __restrict__ vtb,   // [B*16,64,1024] bf16
    const unsigned char*  __restrict__ mask,  // [B,1024] bool
    unsigned short* __restrict__ ob) {        // [B*2048,1024] bf16
  __shared__ __align__(16) unsigned short Pld[32 * 32];
  const int lane = threadIdx.x;
  const int l16 = lane & 15, hi = lane >> 4;
  const int qt = blockIdx.x, h = blockIdx.y, b = blockIdx.z;
  const int q0 = qt * 32;

  const unsigned short* Qb = qb + (size_t)b * 2048 * 1024 + h * 64;
  const unsigned short* Kb = kvb + (size_t)b * 1024 * 2048 + h * 64;
  const unsigned short* Vt = vtb + ((size_t)(b * 16 + h)) * 64 * 1024;
  const unsigned char*  Mb = mask + (size_t)b * 1024;

  v16bf aq[2][2];
#pragma unroll
  for (int ms = 0; ms < 2; ++ms)
#pragma unroll
    for (int ks = 0; ks < 2; ++ks)
      aq[ms][ks] = load_a16x32(Qb, 1024, q0 + ms * 16 + l16, ks * 32, hi);

  v8f acc[2][4] = {};
  float mrun[2][8], lrun[2][8];
#pragma unroll
  for (int ms = 0; ms < 2; ++ms)
#pragma unroll
    for (int r = 0; r < 8; ++r) { mrun[ms][r] = -3.0e38f; lrun[ms][r] = 0.f; }

  for (int j0 = 0; j0 < 1024; j0 += 32) {
    // hint next j-tile into cache while we compute this one
    if (j0 + 32 < 1024) {
      __builtin_prefetch(Kb + (size_t)(j0 + 32 + l16) * 2048, 0, 3);
      __builtin_prefetch(Vt + (size_t)(lane) * 1024 + j0 + 32, 0, 3);
    }
    // S = Q @ K^T   (two 16x16 j-subtiles, d=64 -> two k-steps of 32)
    v8f s[2][2] = {};
#pragma unroll
    for (int js = 0; js < 2; ++js) {
      v16bf bk0 = load_b32x16(Kb, 2048, j0 + js * 16 + l16, 0,  hi);
      v16bf bk1 = load_b32x16(Kb, 2048, j0 + js * 16 + l16, 32, hi);
#pragma unroll
      for (int ms = 0; ms < 2; ++ms) {
        s[ms][js] = WMMA_BF16(aq[ms][0], bk0, s[ms][js]);
        s[ms][js] = WMMA_BF16(aq[ms][1], bk1, s[ms][js]);
      }
    }
    // mask: this lane owns column j0 + js*16 + l16 in the C layout
    float mbias[2];
#pragma unroll
    for (int js = 0; js < 2; ++js)
      mbias[js] = Mb[j0 + js * 16 + l16] ? 0.f : -3.0e38f;
#pragma unroll
    for (int ms = 0; ms < 2; ++ms)
#pragma unroll
      for (int js = 0; js < 2; ++js)
#pragma unroll
        for (int r = 0; r < 8; ++r) s[ms][js][r] += mbias[js];

    // online softmax: row lives in VGPR r across the 16-lane group
    float fac[2][8];
#pragma unroll
    for (int ms = 0; ms < 2; ++ms)
#pragma unroll
      for (int r = 0; r < 8; ++r) {
        float v = fmaxf(s[ms][0][r], s[ms][1][r]);
#pragma unroll
        for (int d = 1; d <= 8; d <<= 1) v = fmaxf(v, __shfl_xor(v, d, 32));
        float mnew = fmaxf(mrun[ms][r], v);
        float f = __expf(mrun[ms][r] - mnew);
        mrun[ms][r] = mnew;
        fac[ms][r] = f;
        float rs = 0.f;
#pragma unroll
        for (int js = 0; js < 2; ++js) {
          float p = __expf(s[ms][js][r] - mnew);
          s[ms][js][r] = p;
          rs += p;
        }
#pragma unroll
        for (int d = 1; d <= 8; d <<= 1) rs += __shfl_xor(rs, d, 32);
        lrun[ms][r] = lrun[ms][r] * f + rs;
      }
    // rescale O
#pragma unroll
    for (int ms = 0; ms < 2; ++ms)
#pragma unroll
      for (int ns = 0; ns < 4; ++ns)
#pragma unroll
        for (int r = 0; r < 8; ++r) acc[ms][ns][r] *= fac[ms][r];

    // P: C-layout -> row-major bf16 in LDS -> reload in A-layout
#pragma unroll
    for (int ms = 0; ms < 2; ++ms)
#pragma unroll
      for (int js = 0; js < 2; ++js)
#pragma unroll
        for (int r = 0; r < 8; ++r)
          Pld[(ms * 16 + r + hi * 8) * 32 + js * 16 + l16] = f2bf(s[ms][js][r]);
    __syncthreads();
    v16bf pa[2];
#pragma unroll
    for (int ms = 0; ms < 2; ++ms)
      pa[ms] = load_a16x32(Pld, 32, ms * 16 + l16, 0, hi);
    __syncthreads();

    // O += P @ V   (V B-operand = contiguous rows of Vt)
#pragma unroll
    for (int ns = 0; ns < 4; ++ns) {
      v16bf bv = load_b32x16(Vt, 1024, ns * 16 + l16, j0, hi);
#pragma unroll
      for (int ms = 0; ms < 2; ++ms)
        acc[ms][ns] = WMMA_BF16(pa[ms], bv, acc[ms][ns]);
    }
  }

  // normalize and store bf16
#pragma unroll
  for (int ms = 0; ms < 2; ++ms)
#pragma unroll
    for (int r = 0; r < 8; ++r) {
      float inv = 1.0f / lrun[ms][r];
      int row = q0 + ms * 16 + r + hi * 8;
#pragma unroll
      for (int ns = 0; ns < 4; ++ns) {
        int col = ns * 16 + l16;
        ob[((size_t)b * 2048 + row) * 1024 + h * 64 + col] =
            f2bf(acc[ms][ns][r] * inv);
      }
    }
}

// ---------------------------------------------------------------------------
// Host-side orchestration
// ---------------------------------------------------------------------------
extern "C" void kernel_launch(void* const* d_in, const int* in_sizes, int n_in,
                              void* d_out, int out_size, void* d_ws, size_t ws_size,
                              hipStream_t stream) {
  const float* x    = (const float*)d_in[0];  // [4,2048,1024]
  const float* ctx  = (const float*)d_in[1];  // [4,1024,1024]
  const unsigned char* mask = (const unsigned char*)d_in[2];  // [4,1024] bool
  const float* Wq   = (const float*)d_in[3];  // [1024,1024]
  const float* Wkv  = (const float*)d_in[4];  // [2048,1024]
  const float* Wout = (const float*)d_in[5];  // [1024,1024]
  const float* bout = (const float*)d_in[6];  // [1024]
  float* out = (float*)d_out;                 // [4,2048,1024] f32

  char* ws = (char*)d_ws;
  const size_t MB = 1024 * 1024;
  unsigned short* xb   = (unsigned short*)(ws + 0);        // 16MB [8192,1024]
  unsigned short* cb   = (unsigned short*)(ws + 16 * MB);  //  8MB [4096,1024]
  unsigned short* wqb  = (unsigned short*)(ws + 24 * MB);  //  2MB (pre-scaled)
  unsigned short* wkvb = (unsigned short*)(ws + 26 * MB);  //  4MB
  unsigned short* wob  = (unsigned short*)(ws + 30 * MB);  //  2MB
  unsigned short* qb   = (unsigned short*)(ws + 32 * MB);  // 16MB [8192,1024]
  unsigned short* kvb  = (unsigned short*)(ws + 48 * MB);  // 16MB [4096,2048]
  unsigned short* vtb  = (unsigned short*)(ws + 64 * MB);  //  8MB [64,64,1024]
  unsigned short* aob  = (unsigned short*)(ws + 72 * MB);  // 16MB [8192,1024]

  const float SCALE = 0.125f;  // 64^-0.5, folded into Wq

  cvt_f32_bf16<<<2048, 256, 0, stream>>>(x,    xb,   (size_t)4 * 2048 * 1024, 1.f);
  cvt_f32_bf16<<<2048, 256, 0, stream>>>(ctx,  cb,   (size_t)4 * 1024 * 1024, 1.f);
  cvt_f32_bf16<<<1024, 256, 0, stream>>>(Wq,   wqb,  (size_t)1024 * 1024, SCALE);
  cvt_f32_bf16<<<1024, 256, 0, stream>>>(Wkv,  wkvb, (size_t)2048 * 1024, 1.f);
  cvt_f32_bf16<<<1024, 256, 0, stream>>>(Wout, wob,  (size_t)1024 * 1024, 1.f);

  // Q = x @ (SCALE*Wq)^T            M=8192 N=1024 K=1024
  gemm_bf16_wmma<false><<<dim3(64, 16), 256, 0, stream>>>(
      xb, wqb, qb, nullptr, 1024, 1024, 1024, 1024);
  // KV = ctx @ Wkv^T                M=4096 N=2048 K=1024
  gemm_bf16_wmma<false><<<dim3(32, 32), 256, 0, stream>>>(
      cb, wkvb, kvb, nullptr, 1024, 1024, 1024, 2048);
  // Vt[b,h,d,j]
  vtrans_kernel<<<16384, 256, 0, stream>>>(kvb, vtb);
  // attention
  attn_fwd_wmma<<<dim3(64, 16, 4), 32, 0, stream>>>(qb, kvb, vtb, mask, aob);
  // out = attn_out @ Wout^T + b     M=8192 N=1024 K=1024, f32 out
  gemm_bf16_wmma<true><<<dim3(64, 16), 256, 0, stream>>>(
      aob, wob, out, bout, 1024, 1024, 1024, 1024);
}
